// SOENModel_56307021251137
// MI455X (gfx1250) — matrix operand order repaired and checked by reading, not compiled, not run on hardware
//
#include <hip/hip_runtime.h>
#include <hip/hip_bf16.h>
#include <math.h>

// ---------------------------------------------------------------------------
// SOEN recurrent dynamics on MI455X (gfx1250, wave32, WMMA).
//   per step:  phi  = s @ (J*mask)^T  + flux_offset + ext
//              phiz = s @ (JZ*mask)^T + flux_offset_Z
//              s   <- clip(s + dt*(gamma * g(phi,s,phiz) - s/tau), -1, 1)
// Compute-bound: bf16 WMMA, fp32 accumulate. B tiles staged through LDS
// (double-buffered); global loads for the next tile overlap the WMMA burst;
// all 16 ds_loads of a k-step are issued before the WMMAs (enforced with
// sched_barrier) so dscnt waits are partial, not wait-0.
// ---------------------------------------------------------------------------

typedef __attribute__((ext_vector_type(16))) __bf16 v16bf;
typedef __attribute__((ext_vector_type(8)))  __bf16 v8bf;
typedef __attribute__((ext_vector_type(8)))  float  v8f;

#define N_TOT   2048
#define N_INP   256
#define N_OUTP  256
#define BATCH   1024
#define ITERS   50
#define DT      0.05f
#define I_B     1.7f

// LDS geometry: per k-step tile = 64 cols x 32 K bf16 per matrix.
// Row stride padded to 80B -> 16-lane b128 reads hit 16 distinct banks.
#define LDS_COL_STRIDE 80
#define LDS_MAT_BYTES  (64 * LDS_COL_STRIDE)   // 5120 B per matrix
#define LDS_BUF_BYTES  (2 * LDS_MAT_BYTES)     // J + Z = 10240 B
#define LDS_TOTAL      (2 * LDS_BUF_BYTES)     // double buffered = 20480 B

union V16 { v16bf v; v8bf h[2]; };

#if defined(__has_builtin)
#if __has_builtin(__builtin_amdgcn_sched_barrier)
#define SCHED_FENCE() __builtin_amdgcn_sched_barrier(0)
#endif
#endif
#ifndef SCHED_FENCE
#define SCHED_FENCE() asm volatile("" ::: "memory")
#endif

// ---------------------------------------------------------------------------
// One-time (per launch) prep: masked weights -> bf16, row-major [n, k].
// ---------------------------------------------------------------------------
__global__ void soen_mask_convert(const float* __restrict__ J,
                                  const float* __restrict__ JZ,
                                  const float* __restrict__ mask,
                                  __bf16* __restrict__ Jmb,
                                  __bf16* __restrict__ JZmb,
                                  int total) {
  int i = blockIdx.x * blockDim.x + threadIdx.x;
  if (i < total) {
    float m = mask[i];
    Jmb[i]  = (__bf16)(J[i]  * m);
    JZmb[i] = (__bf16)(JZ[i] * m);
  }
}

__global__ void soen_init_state(const float* __restrict__ s0,
                                float* __restrict__ sf,
                                __bf16* __restrict__ sb,
                                int total) {
  int i = blockIdx.x * blockDim.x + threadIdx.x;
  if (i < total) {
    float v = s0[i];
    sf[i] = v;
    sb[i] = (__bf16)v;
  }
}

// ---------------------------------------------------------------------------
// One Euler step. Block = 256 threads = 8 waves. Block tile: M=128, N=64.
// Wave w owns rows [m0, m0+16); all waves share the N-strip [n0, n0+64).
// k-loop unrolled x2: even 32-K half uses LDS buf0 + A-frag aA, odd half
// uses buf1 + aB -> no A-frag register copies (no WMMA WAR hazard nops).
// ---------------------------------------------------------------------------
__global__ __launch_bounds__(256)
void soen_step(const __bf16* __restrict__ sb_cur,   // bf16 state  [B, N]
               const float*  __restrict__ sf_cur,   // f32  state  [B, N]
               float*        __restrict__ sf_nxt,
               __bf16*       __restrict__ sb_nxt,
               const __bf16* __restrict__ Jmb,      // [N, K] row-major
               const __bf16* __restrict__ JZmb,     // [N, K] row-major
               const float*  __restrict__ x,        // [B, N_INP]
               const float*  __restrict__ gamma_v,
               const float*  __restrict__ tau_v,
               const float*  __restrict__ fo_v,
               const float*  __restrict__ foz_v) {
  __shared__ __align__(32) char lds[LDS_TOTAL];

  const int tid  = threadIdx.x;
  const int lane = tid & 31;
  const int wave = tid >> 5;
  const int half = lane >> 4;     // 0: lanes 0-15, 1: lanes 16-31
  const int l16  = lane & 15;

  const int m0 = blockIdx.y * 128 + wave * 16;
  const int n0 = blockIdx.x * 64;

  // --- cooperative B-staging role: each thread moves one 32B chunk/k-step ---
  // threads 0..127 -> J tile, 128..255 -> Z tile (wave-uniform split)
  const int sel  = tid >> 7;
  const int sidx = tid & 127;
  const int scol = sidx >> 1;     // 0..63  column within N-strip
  const int skh  = sidx & 1;      // K half (0: k0+0..15, 1: k0+16..31)
  const __bf16* wsrc =
      (sel ? JZmb : Jmb) + (size_t)(n0 + scol) * N_TOT + 16 * skh;
  char* sdst = lds + sel * LDS_MAT_BYTES + scol * LDS_COL_STRIDE + 32 * skh;

  v8f accJ[4], accZ[4];
#pragma unroll
  for (int t = 0; t < 4; ++t) {
    accJ[t] = (v8f)(0.0f);
    accZ[t] = (v8f)(0.0f);
  }

  // A-operand base: lane L holds row m0 + (L&15); K chunks interleave by half.
  const __bf16* aptr = sb_cur + (size_t)(m0 + l16) * N_TOT;
  const int cb0 = l16 * LDS_COL_STRIDE + 16 * half;  // per-lane frag offset

  // One 32-K compute phase: issue ALL 16 ds_loads, hard scheduling fence,
  // then 8 WMMAs -> partial s_wait_dscnt (0xc/0x8/0x4/0x0) instead of
  // wait-0 before every WMMA.
  auto compute = [&](const V16& af, const char* lbase) {
    V16 fj[4], fz[4];
#pragma unroll
    for (int t = 0; t < 4; ++t) {
      const int cb = cb0 + t * 16 * LDS_COL_STRIDE;
      fj[t].h[0] = *(const v8bf*)(lbase + cb);
      fj[t].h[1] = *(const v8bf*)(lbase + cb + 32);
      fz[t].h[0] = *(const v8bf*)(lbase + LDS_MAT_BYTES + cb);
      fz[t].h[1] = *(const v8bf*)(lbase + LDS_MAT_BYTES + cb + 32);
    }
    SCHED_FENCE();
#pragma unroll
    for (int t = 0; t < 4; ++t) {
      accJ[t] = __builtin_amdgcn_wmma_f32_16x16x32_bf16(
          false, af.v, false, fj[t].v, (short)0, accJ[t], false, false);
      accZ[t] = __builtin_amdgcn_wmma_f32_16x16x32_bf16(
          false, af.v, false, fz[t].v, (short)0, accZ[t], false, false);
    }
  };

  // ---- prologue: stage k0 = 0 tile into LDS buf0, preload A frag 0 ----
  {
    V16 st;
    st.h[0] = *(const v8bf*)(wsrc);
    st.h[1] = *(const v8bf*)(wsrc + 8);
    *(v8bf*)(sdst)      = st.h[0];
    *(v8bf*)(sdst + 16) = st.h[1];
  }
  V16 aA, aB;
  aA.h[0] = *(const v8bf*)(aptr + 8 * half);
  aA.h[1] = *(const v8bf*)(aptr + 16 + 8 * half);
  __syncthreads();

  for (int k0 = 0; k0 < N_TOT; k0 += 64) {
    // ---- even half: tile k0 from buf0; prefetch tile k0+32 -> buf1 ----
    {
      const __bf16* wn = wsrc + k0 + 32;          // k0+32 always < N_TOT here
      V16 stg;
      stg.h[0] = *(const v8bf*)(wn);
      stg.h[1] = *(const v8bf*)(wn + 8);
      aB.h[0] = *(const v8bf*)(aptr + k0 + 32 + 8 * half);
      aB.h[1] = *(const v8bf*)(aptr + k0 + 48 + 8 * half);
      compute(aA, lds);                           // buf0
      *(v8bf*)(sdst + LDS_BUF_BYTES)      = stg.h[0];
      *(v8bf*)(sdst + LDS_BUF_BYTES + 16) = stg.h[1];
    }
    __syncthreads();
    // ---- odd half: tile k0+32 from buf1; prefetch tile k0+64 -> buf0 ----
    {
      const bool more = (k0 + 64) < N_TOT;
      V16 stg;
      if (more) {
        const __bf16* wn = wsrc + k0 + 64;
        stg.h[0] = *(const v8bf*)(wn);
        stg.h[1] = *(const v8bf*)(wn + 8);
        aA.h[0] = *(const v8bf*)(aptr + k0 + 64 + 8 * half);
        aA.h[1] = *(const v8bf*)(aptr + k0 + 80 + 8 * half);
      }
      compute(aB, lds + LDS_BUF_BYTES);           // buf1
      if (more) {
        *(v8bf*)(sdst)      = stg.h[0];
        *(v8bf*)(sdst + 16) = stg.h[1];
      }
      __syncthreads();
    }
  }

  // ---- epilogue: C/D layout -> VGPR j, lanes 0-15: (M=j, N=lane);
  //      lanes 16-31: (M=j+8, N=lane-16). Nonlinear Euler update in-register.
#pragma unroll
  for (int t = 0; t < 4; ++t) {
    const int n    = n0 + t * 16 + l16;
    const float fo   = fo_v[n];
    const float foz  = foz_v[n];
    const float gam  = gamma_v[n];
    const float itau = 1.0f / tau_v[n];
#pragma unroll
    for (int j = 0; j < 8; ++j) {
      const int m = m0 + half * 8 + j;
      float phi = accJ[t][j] + fo;
      if (n < N_INP) phi += x[m * N_INP + n];          // external drive
      float phz = accZ[t][j] + foz;
      phi = fminf(0.5f, fmaxf(-0.5f, phi));            // clip phi
      const float phiw = phi - rintf(phi);             // period-1 wrap (RNE)
      const float sv = sf_cur[(size_t)m * N_TOT + n];
      const float g  = tanhf(I_B * phiw) * (1.0f - sv) * (1.0f + tanhf(phz));
      float sn = sv + DT * (gam * g - sv * itau);
      sn = fminf(1.0f, fmaxf(-1.0f, sn));              // clip state
      sf_nxt[(size_t)m * N_TOT + n] = sn;
      sb_nxt[(size_t)m * N_TOT + n] = (__bf16)sn;
    }
  }
}

// ---------------------------------------------------------------------------
// Gather the output-node states: s_final[:, N-N_OUT:]
// ---------------------------------------------------------------------------
__global__ void soen_extract(const float* __restrict__ sf, float* __restrict__ out,
                             int total) {
  int i = blockIdx.x * blockDim.x + threadIdx.x;
  if (i < total) {
    int b = i >> 8;           // N_OUTP == 256
    int j = i & 255;
    out[i] = sf[(size_t)b * N_TOT + (N_TOT - N_OUTP) + j];
  }
}

// ---------------------------------------------------------------------------
// inputs (setup_inputs order):
//  0:x [B,256]  1:s0 [B,N]  2:J [N,N]  3:JZ [N,N]  4:gamma [N]  5:tau [N]
//  6:flux_offset [N]  7:flux_offset_Z [N]  8:mask [N,N]
// ---------------------------------------------------------------------------
extern "C" void kernel_launch(void* const* d_in, const int* in_sizes, int n_in,
                              void* d_out, int out_size, void* d_ws, size_t ws_size,
                              hipStream_t stream) {
  const float* x     = (const float*)d_in[0];
  const float* s0    = (const float*)d_in[1];
  const float* J     = (const float*)d_in[2];
  const float* JZ    = (const float*)d_in[3];
  const float* gamma = (const float*)d_in[4];
  const float* tau   = (const float*)d_in[5];
  const float* fo    = (const float*)d_in[6];
  const float* foz   = (const float*)d_in[7];
  const float* mask  = (const float*)d_in[8];
  float* out = (float*)d_out;

  // Workspace carve-up (40 MB total).
  char* ws = (char*)d_ws;
  const size_t szW  = (size_t)N_TOT * N_TOT * sizeof(__bf16);   // 8 MB
  const size_t szSf = (size_t)BATCH * N_TOT * sizeof(float);    // 8 MB
  const size_t szSb = (size_t)BATCH * N_TOT * sizeof(__bf16);   // 4 MB
  __bf16* Jmb   = (__bf16*)(ws);
  __bf16* JZmb  = (__bf16*)(ws + szW);
  float*  sf[2] = { (float*)(ws + 2 * szW),
                    (float*)(ws + 2 * szW + szSf) };
  __bf16* sb[2] = { (__bf16*)(ws + 2 * szW + 2 * szSf),
                    (__bf16*)(ws + 2 * szW + 2 * szSf + szSb) };

  // 1) mask + bf16-convert weights (every call: no cross-call state allowed)
  {
    const int total = N_TOT * N_TOT;
    soen_mask_convert<<<total / 256, 256, 0, stream>>>(J, JZ, mask, Jmb, JZmb, total);
  }
  // 2) initialize state buffers from s0
  {
    const int total = BATCH * N_TOT;
    soen_init_state<<<total / 256, 256, 0, stream>>>(s0, sf[0], sb[0], total);
  }
  // 3) 50 sequential Euler steps, ping-ponging the state
  dim3 grid(N_TOT / 64, BATCH / 128);   // 32 x 8 blocks
  int cur = 0;
  for (int it = 0; it < ITERS; ++it) {
    soen_step<<<grid, 256, 0, stream>>>(sb[cur], sf[cur], sf[cur ^ 1], sb[cur ^ 1],
                                        Jmb, JZmb, x, gamma, tau, fo, foz);
    cur ^= 1;
  }
  // 4) emit s_final[:, N-N_OUT:]
  {
    const int total = BATCH * N_OUTP;
    soen_extract<<<total / 256, 256, 0, stream>>>(sf[cur], out, total);
  }
}
